// MultiHeadAttention_24472723652682
// MI455X (gfx1250) — compile-verified, matrix operand order checked by reading, not run on hardware
//
#include <hip/hip_runtime.h>
#include <hip/hip_bf16.h>
#include <math.h>

// ---------------------------------------------------------------------------
// MI455X (gfx1250) multi-head attention, bf16 WMMA pipeline.
//   ~103 GFLOP vs ~10us of HBM time at 23.3 TB/s -> compute bound on the
//   matrix units. All GEMM-shaped work goes through v_wmma_f32_16x16x32_bf16.
//   Operands are converted to bf16 once; GEMM tiles are staged into LDS with
//   CDNA5 async copies (ASYNCcnt) and the Tensor Data Mover (TENSORcnt).
// ---------------------------------------------------------------------------

typedef __bf16 bf16_t;
typedef unsigned int  u32;
typedef unsigned long long u64;
typedef __attribute__((ext_vector_type(16))) __bf16 v16bf;
typedef __attribute__((ext_vector_type(8)))  __bf16 v8bf;
typedef __attribute__((ext_vector_type(8)))  float  v8f;
typedef __attribute__((ext_vector_type(4)))  u32    u32x4;
typedef __attribute__((ext_vector_type(8)))  int    i32x8;
typedef __attribute__((ext_vector_type(4)))  int    i32x4;

__device__ __forceinline__ v16bf make_frag(v8bf lo, v8bf hi) {
    v16bf r;
#pragma unroll
    for (int e = 0; e < 8; ++e) { r[e] = lo[e]; r[e + 8] = hi[e]; }
    return r;
}

// ---------------------------------------------------------------------------
// Kernel 0: one-time f32 -> bf16 conversion (halves GEMM stream traffic).
// ---------------------------------------------------------------------------
__global__ void f32_to_bf16_kernel(const float* __restrict__ in,
                                   bf16_t* __restrict__ out, size_t n) {
    const size_t i = ((size_t)blockIdx.x * blockDim.x + threadIdx.x) * 4;
    if (i + 3 < n) {
        float4 v = *(const float4*)(in + i);
        out[i]     = (bf16_t)v.x;
        out[i + 1] = (bf16_t)v.y;
        out[i + 2] = (bf16_t)v.z;
        out[i + 3] = (bf16_t)v.w;
    }
}

// ---------------------------------------------------------------------------
// Kernel 1/4: C[M,N] = A[M,K] @ B[K,N] + bias (bf16 in, f32 acc).
// Block tile 64x64, 8 waves x (16x32 out slab). A tile staged with async
// global->LDS b128 copies; B tile staged with one TDM descriptor per K step
// (D# padding produces the bank-staggered LDS stride in hardware).
// ---------------------------------------------------------------------------
template <bool BF16_OUT>
__global__ __launch_bounds__(256) void gemm_bias_wmma(
    const bf16_t* __restrict__ A, const bf16_t* __restrict__ Bm,
    const float* __restrict__ bias, void* __restrict__ Cout,
    int M, int N, int K)
{
    constexpr int AST = 40;   // A LDS row stride (bf16): 80B, 16B-aligned
    constexpr int BST = 72;   // B LDS row stride: 64 data + 8 pad elems (TDM pad)
    __shared__ bf16_t As[64 * AST];
    __shared__ bf16_t Bs[32 * BST];

    const int tid  = threadIdx.x;
    const int lane = tid & 31;
    const int wave = tid >> 5;
    const int m0 = blockIdx.y * 64;
    const int n0 = blockIdx.x * 64;
    const int wm = (wave & 3) * 16;
    const int wn = (wave >> 2) * 32;
    const int r  = lane & 15;
    const int s  = (lane >> 4) * 8;
    const int rbase = (lane >> 4) * 8;

    // A staging map: 64 rows x 4 chunks of 8 bf16 (16B) = 256 lanes
    const int arow = tid >> 2, achk = (tid & 3) * 8;
    const u32 ldsA = (u32)(uintptr_t)(As + arow * AST + achk);
    const u32 ldsB = (u32)(uintptr_t)(Bs);

    // TDM D# group1: 2D tile 64(x) x 32(y) of 2-byte elems, row stride N,
    // pad_enable, pad_interval = 32 DWORDS (one 64-elem row), pad_amount = 4 DW.
    i32x8 g1;
    g1[0] = (1 << 16) | (1 << 20) | (4 << 22) | (3 << 25);
    g1[1] = (int)(((u32)N & 0xFFFFu) << 16);            // tensor_dim0[15:0]
    g1[2] = (int)(((u32)N >> 16) | (((u32)K & 0xFFFFu) << 16)); // dim0 hi | dim1 lo
    g1[3] = (int)(((u32)K >> 16) | (64u << 16));        // dim1 hi | tile_dim0 = 64
    g1[4] = 32;                                         // tile_dim1 = 32
    g1[5] = N;                                          // tensor_dim0_stride lo
    g1[6] = 0;
    g1[7] = 0;
    const i32x4 gz4 = {0, 0, 0, 0};
    const i32x8 gz8 = {0, 0, 0, 0, 0, 0, 0, 0};

    v8f acc0 = {}, acc1 = {};

    for (int k0 = 0; k0 < K; k0 += 32) {
        // ---- B tile via Tensor Data Mover (issued once, by wave 0) ----
        if (wave == 0) {
            const u64 ga = (u64)(uintptr_t)(Bm + (size_t)k0 * N + n0);
            u32x4 g0;
            g0[0] = 1u;                                   // count = 1 descriptor
            g0[1] = ldsB;                                 // lds_addr
            g0[2] = (u32)ga;                              // global_addr[31:0]
            g0[3] = ((u32)(ga >> 32) & 0x01FFFFFFu) | 0x80000000u; // addr hi | type=2
            __builtin_amdgcn_tensor_load_to_lds(g0, g1, gz4, gz4, gz8, 0);
        }
        // ---- A tile via async global->LDS copies (16B per lane) ----
        {
            const u64 ga = (u64)(uintptr_t)(A + (size_t)(m0 + arow) * K + k0 + achk);
            asm volatile("global_load_async_to_lds_b128 %0, %1, off"
                         :: "v"(ldsA), "v"(ga) : "memory");
        }
        asm volatile("s_wait_asynccnt 0x0" ::: "memory");
        __builtin_amdgcn_s_wait_tensorcnt(0);
        __syncthreads();

        // A fragment: lane = row (wm+r), K phase s..s+7 / 16+s..16+s+7
        v16bf afrag = make_frag(*(const v8bf*)(As + (wm + r) * AST + s),
                                *(const v8bf*)(As + (wm + r) * AST + 16 + s));
        // B fragments: lane = col, gathered across padded LDS rows
        v16bf bf0, bf1;
#pragma unroll
        for (int e = 0; e < 8; ++e) {
            bf0[e]     = Bs[(s + e) * BST      + wn + r];
            bf0[e + 8] = Bs[(16 + s + e) * BST + wn + r];
            bf1[e]     = Bs[(s + e) * BST      + wn + 16 + r];
            bf1[e + 8] = Bs[(16 + s + e) * BST + wn + 16 + r];
        }
        acc0 = __builtin_amdgcn_wmma_f32_16x16x32_bf16(false, afrag, false, bf0,
                                                       (short)0, acc0, false, false);
        acc1 = __builtin_amdgcn_wmma_f32_16x16x32_bf16(false, afrag, false, bf1,
                                                       (short)0, acc1, false, false);
        __syncthreads();
    }

    const int c0 = n0 + wn + r, c1 = c0 + 16;
    const float bv0 = bias[c0], bv1 = bias[c1];
#pragma unroll
    for (int i = 0; i < 8; ++i) {
        const int row = m0 + wm + rbase + i;
        const float v0 = acc0[i] + bv0;
        const float v1 = acc1[i] + bv1;
        if (BF16_OUT) {
            bf16_t* o = (bf16_t*)Cout;
            o[(size_t)row * N + c0] = (bf16_t)v0;
            o[(size_t)row * N + c1] = (bf16_t)v1;
        } else {
            float* o = (float*)Cout;
            o[(size_t)row * N + c0] = v0;
            o[(size_t)row * N + c1] = v1;
        }
    }
}

// ---------------------------------------------------------------------------
// Kernel 2: RoPE + head split.  qkv[B,T,3C] bf16 ->
//   Qr,Kr [B,H,T,64] bf16 (row-major: contiguous fragment loads)
//   Vt    [B,H,64,T] bf16 (transposed: P*V B-fragments contiguous along K)
// ---------------------------------------------------------------------------
__global__ void rope_split_kernel(const bf16_t* __restrict__ qkv,
                                  bf16_t* __restrict__ Qr,
                                  bf16_t* __restrict__ Kr,
                                  bf16_t* __restrict__ Vt,
                                  int B, int T)
{
    const size_t total = (size_t)B * T * 16 * 64;
    const size_t idx = (size_t)blockIdx.x * blockDim.x + threadIdx.x;
    if (idx >= total) return;
    const int d = (int)(idx & 63);
    const int h = (int)((idx >> 6) & 15);
    const int t = (int)((idx >> 10) & (size_t)(T - 1));   // T = 2048 (pow2)
    const int b = (int)(idx >> 21);

    const bf16_t* base = qkv + (size_t)(b * T + t) * 3072;
    const int col = h * 64 + d;
    const float qv = (float)base[col];
    const float kv = (float)base[1024 + col];
    const float vv = (float)base[2048 + col];

    const int p = d & 31;
    const float inv = powf(10000.0f, -(float)(2 * p) / 64.0f);
    const float ang = (float)t * inv;
    const float cs = __cosf(ang), sn = __sinf(ang);
    const int dpart = (d < 32) ? d + 32 : d - 32;
    const float qp = (float)base[h * 64 + dpart];
    const float kp = (float)base[1024 + h * 64 + dpart];
    const float qo = qv * cs + ((d < 32) ? -qp : qp) * sn;
    const float ko = kv * cs + ((d < 32) ? -kp : kp) * sn;

    const size_t bh = (size_t)(b * 16 + h);
    Qr[(bh * T + t) * 64 + d] = (bf16_t)qo;
    Kr[(bh * T + t) * 64 + d] = (bf16_t)ko;
    Vt[(bh * 64 + d) * T + t] = (bf16_t)vv;
}

// ---------------------------------------------------------------------------
// Kernel 3: causal flash attention. One wave per 16-query tile, 32 keys/iter,
// online softmax, 8 WMMAs per iteration (4 for S=QK^T, 4 for O+=PV).
// Output written in [B,T,C] bf16 so the final projection streams bf16.
// ---------------------------------------------------------------------------
__global__ __launch_bounds__(256) void flash_attn_wmma(
    const bf16_t* __restrict__ Q, const bf16_t* __restrict__ Kr,
    const bf16_t* __restrict__ Vt, bf16_t* __restrict__ O,
    int T, int H)
{
    constexpr int PST = 40;
    __shared__ bf16_t Pl[8 * 16 * PST];         // wave-private slices

    const int lane = threadIdx.x & 31;
    const int wave = threadIdx.x >> 5;
    const int nqb  = T / 128;
    const int bh   = blockIdx.x / nqb;
    const int qblk = blockIdx.x % nqb;
    const int h = bh % H, b = bh / H;
    const int q0 = qblk * 128 + wave * 16;

    const bf16_t* Qb = Q  + (size_t)bh * T * 64;
    const bf16_t* Kb = Kr + (size_t)bh * T * 64;
    const bf16_t* Vb = Vt + (size_t)bh * 64 * T;
    bf16_t* Pw = Pl + wave * 16 * PST;

    const int r = lane & 15;
    const int s = (lane >> 4) * 8;
    const int rbase = (lane >> 4) * 8;

    v16bf qa0, qa1;
    {
        const bf16_t* qrow = Qb + (size_t)(q0 + r) * 64;
        qa0 = make_frag(*(const v8bf*)(qrow + s),      *(const v8bf*)(qrow + 16 + s));
        qa1 = make_frag(*(const v8bf*)(qrow + 32 + s), *(const v8bf*)(qrow + 48 + s));
    }

    v8f ov[4] = {{}, {}, {}, {}};
    float mrow[8], lrow[8];
#pragma unroll
    for (int i = 0; i < 8; ++i) { mrow[i] = -3.0e38f; lrow[i] = 0.0f; }

    for (int kt0 = 0; kt0 <= q0 + 15; kt0 += 32) {
        v8f s0 = {}, s1 = {};
        {
            const bf16_t* krow0 = Kb + (size_t)(kt0 + r) * 64;
            const bf16_t* krow1 = Kb + (size_t)(kt0 + 16 + r) * 64;
            v16bf kb00 = make_frag(*(const v8bf*)(krow0 + s),      *(const v8bf*)(krow0 + 16 + s));
            v16bf kb01 = make_frag(*(const v8bf*)(krow0 + 32 + s), *(const v8bf*)(krow0 + 48 + s));
            v16bf kb10 = make_frag(*(const v8bf*)(krow1 + s),      *(const v8bf*)(krow1 + 16 + s));
            v16bf kb11 = make_frag(*(const v8bf*)(krow1 + 32 + s), *(const v8bf*)(krow1 + 48 + s));
            s0 = __builtin_amdgcn_wmma_f32_16x16x32_bf16(false, qa0, false, kb00, (short)0, s0, false, false);
            s0 = __builtin_amdgcn_wmma_f32_16x16x32_bf16(false, qa1, false, kb01, (short)0, s0, false, false);
            s1 = __builtin_amdgcn_wmma_f32_16x16x32_bf16(false, qa0, false, kb10, (short)0, s1, false, false);
            s1 = __builtin_amdgcn_wmma_f32_16x16x32_bf16(false, qa1, false, kb11, (short)0, s1, false, false);
        }

        const float sc = 0.125f;                     // 1/sqrt(64)
        float p0[8], p1[8];
#pragma unroll
        for (int i = 0; i < 8; ++i) {
            const int qg = q0 + rbase + i;
            p0[i] = ((kt0 + r)      <= qg) ? s0[i] * sc : -3.0e38f;
            p1[i] = ((kt0 + 16 + r) <= qg) ? s1[i] * sc : -3.0e38f;
        }

#pragma unroll
        for (int i = 0; i < 8; ++i) {
            float mx = fmaxf(p0[i], p1[i]);
            mx = fmaxf(mx, __shfl_xor(mx, 1, 32));
            mx = fmaxf(mx, __shfl_xor(mx, 2, 32));
            mx = fmaxf(mx, __shfl_xor(mx, 4, 32));
            mx = fmaxf(mx, __shfl_xor(mx, 8, 32));
            const float mnew  = fmaxf(mrow[i], mx);
            const float alpha = __expf(mrow[i] - mnew);
            const float e0 = __expf(p0[i] - mnew);
            const float e1 = __expf(p1[i] - mnew);
            float sum = e0 + e1;
            sum += __shfl_xor(sum, 1, 32);
            sum += __shfl_xor(sum, 2, 32);
            sum += __shfl_xor(sum, 4, 32);
            sum += __shfl_xor(sum, 8, 32);
            lrow[i] = lrow[i] * alpha + sum;
            mrow[i] = mnew;
#pragma unroll
            for (int dt = 0; dt < 4; ++dt) ov[dt][i] *= alpha;
            Pw[(rbase + i) * PST + r]      = (bf16_t)e0;
            Pw[(rbase + i) * PST + 16 + r] = (bf16_t)e1;
        }

        v16bf pa = make_frag(*(const v8bf*)(Pw + r * PST + s),
                             *(const v8bf*)(Pw + r * PST + 16 + s));
#pragma unroll
        for (int dt = 0; dt < 4; ++dt) {
            const bf16_t* vrow = Vb + (size_t)(16 * dt + r) * T + kt0;
            v16bf vb = make_frag(*(const v8bf*)(vrow + s), *(const v8bf*)(vrow + 16 + s));
            ov[dt] = __builtin_amdgcn_wmma_f32_16x16x32_bf16(false, pa, false, vb,
                                                             (short)0, ov[dt], false, false);
        }
    }

#pragma unroll
    for (int i = 0; i < 8; ++i) {
        const float il = 1.0f / lrow[i];
        const int row = q0 + rbase + i;
        bf16_t* orow = O + ((size_t)(b * T + row)) * 1024 + h * 64;
        orow[r]      = (bf16_t)(ov[0][i] * il);
        orow[16 + r] = (bf16_t)(ov[1][i] * il);
        orow[32 + r] = (bf16_t)(ov[2][i] * il);
        orow[48 + r] = (bf16_t)(ov[3][i] * il);
    }
}

// ---------------------------------------------------------------------------
// Host launcher
// ---------------------------------------------------------------------------
extern "C" void kernel_launch(void* const* d_in, const int* in_sizes, int n_in,
                              void* d_out, int out_size, void* d_ws, size_t ws_size,
                              hipStream_t stream) {
    const float* x     = (const float*)d_in[0];
    const float* w_qkv = (const float*)d_in[1];
    const float* b_qkv = (const float*)d_in[2];
    const float* w_out = (const float*)d_in[3];
    const float* b_out = (const float*)d_in[4];
    float* out = (float*)d_out;

    const int B = 4, T = 2048, C = 1024, H = 16;
    const int M = B * T;  // 8192 tokens

    bf16_t* ws = (bf16_t*)d_ws;
    bf16_t* xb    = ws;                         // M*C
    bf16_t* wqb   = xb  + (size_t)M * C;        // C*3C
    bf16_t* wob   = wqb + (size_t)C * 3 * C;    // C*C
    bf16_t* qkv   = wob + (size_t)C * C;        // M*3C
    bf16_t* Qr    = qkv + (size_t)M * 3 * C;    // M*C
    bf16_t* Kr    = Qr  + (size_t)M * C;
    bf16_t* Vt    = Kr  + (size_t)M * C;
    bf16_t* attnb = Vt  + (size_t)M * C;        // M*C

    // 0) one-time bf16 conversion of activations + weights
    {
        const size_t nx = (size_t)M * C, nwq = (size_t)C * 3 * C, nwo = (size_t)C * C;
        f32_to_bf16_kernel<<<(unsigned)((nx / 4 + 255) / 256), 256, 0, stream>>>(x, xb, nx);
        f32_to_bf16_kernel<<<(unsigned)((nwq / 4 + 255) / 256), 256, 0, stream>>>(w_qkv, wqb, nwq);
        f32_to_bf16_kernel<<<(unsigned)((nwo / 4 + 255) / 256), 256, 0, stream>>>(w_out, wob, nwo);
    }

    // 1) qkv = x @ w_qkv + b_qkv   (bf16 out)
    dim3 g1(3 * C / 64, M / 64);
    gemm_bias_wmma<true><<<g1, 256, 0, stream>>>(xb, wqb, b_qkv, qkv, M, 3 * C, C);

    // 2) RoPE + head split / transpose
    const size_t tot = (size_t)M * C;
    rope_split_kernel<<<(unsigned)((tot + 255) / 256), 256, 0, stream>>>(qkv, Qr, Kr, Vt, B, T);

    // 3) causal flash attention -> attnb [B,T,C] bf16
    flash_attn_wmma<<<B * H * (T / 128), 256, 0, stream>>>(Qr, Kr, Vt, attnb, T, H);

    // 4) out = attn @ w_out + b_out  (f32 out)
    dim3 g2(C / 64, M / 64);
    gemm_bias_wmma<false><<<g2, 256, 0, stream>>>(attnb, wob, b_out, out, M, C, C);
}
